// prob_calc_added_layers_5188320494012
// MI455X (gfx1250) — matrix operand order repaired and checked by reading, not compiled, not run on hardware
//
#include <hip/hip_runtime.h>
#include <stdint.h>

// ---------------------------------------------------------------------------
// CDNA5 (gfx1250) WMMA types / helpers.  Wave32 only.
// ---------------------------------------------------------------------------
typedef __attribute__((ext_vector_type(16))) __bf16 v16bf;
typedef __attribute__((ext_vector_type(8)))  float  v8f;

#define WMMA_BF16(a, b, c)                                                     \
  __builtin_amdgcn_wmma_f32_16x16x32_bf16(false, (a), false, (b), (short)0,    \
                                          (c), false, false)

__device__ __forceinline__ int lane_id() { return threadIdx.x & 31; }

// A fragment (16x32 bf16) from a row-major f32 source (LDS).
// 16-bit A 16x32 layout (contiguous k order, matches B packing):
//   lanes 0-15 : row M=lane,    elems 0..15 = K 0..15
//   lanes16-31 : row M=lane-16, elems 0..15 = K 16..31
__device__ __forceinline__ v16bf frag_a_f32(const float* A, int lda, int kt) {
  const int l = lane_id(), h = l >> 4, m = l & 15;
  const float* row = A + m * lda + kt * 32;
  v16bf a;
#pragma unroll
  for (int j = 0; j < 8; ++j) a[j] = (__bf16)row[h * 16 + j];
#pragma unroll
  for (int j = 0; j < 8; ++j) a[8 + j] = (__bf16)row[h * 16 + 8 + j];
  return a;
}

// A fragment where only matrix row `row` is non-zero; its data is the f32
// vector vec[kt*32 .. kt*32+31].  (Used to phrase mat-vecs as WMMA GEMMs.)
__device__ __forceinline__ v16bf frag_a_onerow(const float* vec, int kt,
                                               int row) {
  const int l = lane_id(), h = l >> 4, m = l & 15;
  const bool on = (m == row);
  const float* p = vec + kt * 32;
  v16bf a;
#pragma unroll
  for (int j = 0; j < 8; ++j) a[j] = on ? (__bf16)p[h * 16 + j] : (__bf16)0.f;
#pragma unroll
  for (int j = 0; j < 8; ++j)
    a[8 + j] = on ? (__bf16)p[h * 16 + 8 + j] : (__bf16)0.f;
  return a;
}

// A fragment from bf16 LDS [8][128] (rows 8..15 are zero).
__device__ __forceinline__ v16bf frag_a_rows8(const __bf16* A, int kt) {
  const int l = lane_id(), h = l >> 4, m = l & 15;
  const bool on = (m < 8);
  const __bf16* row = A + (m & 7) * 128 + kt * 32;
  v16bf a;
#pragma unroll
  for (int j = 0; j < 8; ++j) a[j] = on ? row[h * 16 + j] : (__bf16)0.f;
#pragma unroll
  for (int j = 0; j < 8; ++j)
    a[8 + j] = on ? row[h * 16 + 8 + j] : (__bf16)0.f;
  return a;
}

// Fragment-major packed tile load: tile = [32 lanes][16 bf16] contiguous
// (two global_load_b128 per lane).
__device__ __forceinline__ v16bf frag_pk(const __bf16* P, size_t tile) {
  return *(const v16bf*)(P + (tile * 32 + (size_t)lane_id()) * 16);
}

// B fragment for B = X^T where X is a row-major bf16 [N x ldx] tensor:
// b[e] = B[kt*32 + h*16 + e][nt*16 + m] = X[nt*16+m][kt*32+h*16+e]
// -> one contiguous 32-byte load per lane.
__device__ __forceinline__ v16bf frag_bT(const __bf16* X, int ldx, int nt,
                                         int kt) {
  const int l = lane_id(), h = l >> 4, m = l & 15;
  return *(const v16bf*)(X + (size_t)(nt * 16 + m) * ldx + kt * 32 + h * 16);
}

// C/D f32 16x16: c[r] -> row 8*(lane/16)+r, col lane%16  (ISA 7.12.2)
__device__ __forceinline__ void store_c(float* C, int ldc, int nt, v8f c) {
  const int l = lane_id(), h = l >> 4, m = l & 15;
#pragma unroll
  for (int r = 0; r < 8; ++r) C[(h * 8 + r) * ldc + nt * 16 + m] = c[r];
}

// 16x16 C tile accumulate: A (16 x KT*32 f32 in LDS) x packed weight, col nt.
template <int KT>
__device__ __forceinline__ v8f gemm_acc(v8f c, const float* A, int lda,
                                        const __bf16* Wpk, int nt) {
#pragma unroll
  for (int kt = 0; kt < KT; ++kt) {
    v16bf a = frag_a_f32(A, lda, kt);
    v16bf b = frag_pk(Wpk, (size_t)kt * 16 + nt);
    c = WMMA_BF16(a, b, c);
  }
  return c;
}

// ---------------------------------------------------------------------------
// Pack row-major f32 [K x N] weight into B-fragment bf16 tiles.
// ---------------------------------------------------------------------------
__global__ void k_pack_b(const float* __restrict__ W, __bf16* __restrict__ out,
                         int K, int N) {
  const int NT = N >> 4, KT = K >> 5;
  const int idx = blockIdx.x * blockDim.x + threadIdx.x;
  if (idx >= KT * NT * 32) return;
  const int l = idx & 31, h = l >> 4, m = l & 15;
  const int t = idx >> 5, nt = t % NT, kt = t / NT;
  __bf16* o = out + (size_t)idx * 16;
  const float* src = W + (size_t)(kt * 32 + h * 16) * N + nt * 16 + m;
#pragma unroll
  for (int e = 0; e < 16; ++e) o[e] = (__bf16)src[(size_t)e * N];
}

// Pack row-major f32 [M x K] into A-fragment bf16 tiles (tile = rt*KT + kt).
__global__ void k_pack_a(const float* __restrict__ A, __bf16* __restrict__ out,
                         int K, int MT) {
  const int KT = K >> 5;
  const long long idx = (long long)blockIdx.x * blockDim.x + threadIdx.x;
  const long long t = idx >> 5;
  if (t >= (long long)MT * KT) return;
  const int l = (int)(idx & 31), h = l >> 4, m = l & 15;
  const int kt = (int)(t % KT);
  const long long rt = t / KT;
  const float* row = A + (rt * 16 + m) * (long long)K + kt * 32;
  __bf16* o = out + idx * 16;
#pragma unroll
  for (int j = 0; j < 8; ++j) o[j] = (__bf16)row[h * 16 + j];
#pragma unroll
  for (int j = 0; j < 8; ++j) o[8 + j] = (__bf16)row[h * 16 + 8 + j];
}

// ---------------------------------------------------------------------------
// K = E@Wk (bf16 row-major), Vt = (E@Wv)^T per batch (bf16 [b][d][n]),
// LK = E@Wlk (bf16 row-major).  M=131072, N=K=256, bf16 WMMA, f32 accum.
// grid = (8192 row tiles, 3); 8 waves x two 16x16 column tiles.
// ---------------------------------------------------------------------------
__global__ __launch_bounds__(256) void k_gemm_kvl(
    const __bf16* __restrict__ Epk, const __bf16* __restrict__ Wkpk,
    const __bf16* __restrict__ Wvpk, const __bf16* __restrict__ Wlpk,
    __bf16* __restrict__ K16, __bf16* __restrict__ Vt16,
    __bf16* __restrict__ L16) {
  const int rt = blockIdx.x, y = blockIdx.y;
  const __bf16* Wpk = (y == 0) ? Wkpk : (y == 1) ? Wvpk : Wlpk;
  __bf16* Og = (y == 0) ? K16 : (y == 1) ? Vt16 : L16;
  const int wave = threadIdx.x >> 5, lane = threadIdx.x & 31;
  const int h = lane >> 4, m = lane & 15;
  v16bf afr[8];
#pragma unroll
  for (int kt = 0; kt < 8; ++kt) afr[kt] = frag_pk(Epk, (size_t)rt * 8 + kt);
#pragma unroll
  for (int rep = 0; rep < 2; ++rep) {
    const int nt = wave + rep * 8;
    v8f c;
#pragma unroll
    for (int r = 0; r < 8; ++r) c[r] = 0.f;
#pragma unroll
    for (int kt = 0; kt < 8; ++kt)
      c = WMMA_BF16(afr[kt], frag_pk(Wpk, (size_t)kt * 16 + nt), c);
    if (y == 1) {  // transposed per-batch store: Vt[b][d][n]
      const int g0 = rt * 16 + h * 8;      // 16-row tile never crosses batch
      const int b = g0 >> 7, n0 = g0 & 127;
      __bf16* o = Og + (size_t)b * (256 * 128) + (size_t)(nt * 16 + m) * 128 +
                  n0;
#pragma unroll
      for (int r = 0; r < 8; ++r) o[r] = (__bf16)c[r];
    } else {  // row-major bf16 [B*N][256]
#pragma unroll
      for (int r = 0; r < 8; ++r)
        Og[(size_t)(rt * 16 + h * 8 + r) * 256 + nt * 16 + m] = (__bf16)c[r];
    }
  }
}

// ---------------------------------------------------------------------------
// Persistent decode: 64 blocks x 512 threads (16 wave32).  Each block owns
// 16 batch elements for all 128 greedy steps.  All matmuls (weight GEMMs,
// compat, heads, pointer logits) go through bf16 WMMA with f32 accumulate.
// ---------------------------------------------------------------------------
static constexpr unsigned DEC_SMEM_WORDS =
    4096 /*gl*/ + 8192 /*ctx*/ + 4096 /*q*/ + 4096 /*h1*/ + 4096 /*q2*/ +
    4096 /*hd*/ + 8192 /*attn bf16*/ + 2048 /*logits*/ + 64 /*vis*/ +
    48 /*first/prev/sel*/ + 32 /*ll,cost*/ + 64 /*fco,pco*/;
static constexpr size_t DEC_SMEM_BYTES = (size_t)DEC_SMEM_WORDS * 4;

__global__ __launch_bounds__(512) void k_decode(
    const float* __restrict__ coords, const float* __restrict__ E,
    const float* __restrict__ Wph, const float* __restrict__ b1g,
    const float* __restrict__ b2g, const __bf16* __restrict__ K16,
    const __bf16* __restrict__ Vt16, const __bf16* __restrict__ L16,
    const __bf16* __restrict__ Wcpk, const __bf16* __restrict__ Wspk,
    const __bf16* __restrict__ W1pk, const __bf16* __restrict__ W2pk,
    const __bf16* __restrict__ Wopk, float* __restrict__ out) {
  extern __shared__ float sm[];
  float* s_gl = sm;             // [16][256] glimpse (f32)
  float* s_ctx = s_gl + 4096;   // [16][512] step context
  float* s_q = s_ctx + 8192;    // [16][256]
  float* s_h1 = s_q + 4096;     // [16][256]
  float* s_q2 = s_h1 + 4096;    // [16][256] transformed query
  float* s_hd = s_q2 + 4096;    // [16][256] concat heads
  __bf16* s_at = (__bf16*)(s_hd + 4096);     // [16][8][128] attn weights
  float* s_lg = (float*)(s_at + 16 * 1024);  // [16][128] logits
  unsigned* s_vis = (unsigned*)(s_lg + 2048);
  int* s_first = (int*)(s_vis + 64);
  int* s_prev = s_first + 16;
  int* s_sel = s_prev + 16;
  float* s_ll = (float*)(s_sel + 16);
  float* s_cost = s_ll + 16;
  float* s_fco = s_cost + 16;  // [16][2]
  float* s_pco = s_fco + 32;   // [16][2]

  const int tid = threadIdx.x, wave = tid >> 5, lane = tid & 31;
  const int h16 = lane >> 4, m16 = lane & 15;
  const int bb0 = blockIdx.x * 16;

  // glimpse0 = mean over nodes (f32, exact like reference)
  for (int o = tid; o < 16 * 256; o += 512) {
    const int b = o >> 8, d = o & 255;
    const float* e = E + ((size_t)(bb0 + b) * 128) * 256 + d;
    float s = 0.f;
    for (int n = 0; n < 128; ++n) s += e[(size_t)n * 256];
    s_gl[o] = s * (1.f / 128.f);
  }
  if (tid < 64) s_vis[tid] = 0u;
  if (tid < 16) {
    s_first[tid] = 0; s_prev[tid] = 0; s_sel[tid] = 0;
    s_ll[tid] = 0.f; s_cost[tid] = 0.f;
  }
  __syncthreads();

  for (int step = 0; step < 128; ++step) {
    // ---- step context: placeholder at step 0, else [first_emb, prev_emb]
    for (int o = tid; o < 16 * 512; o += 512) {
      const int b = o >> 9, d = o & 511;
      float v;
      if (step == 0) {
        v = Wph[d];
      } else {
        const int node = (d < 256) ? s_first[b] : s_prev[b];
        v = E[((size_t)(bb0 + b) * 128 + node) * 256 + (d & 255)];
      }
      s_ctx[o] = v;
    }
    __syncthreads();

    // ---- q = glimpse @ Wc + ctx @ Ws   (wave = column tile)
    {
      v8f c;
#pragma unroll
      for (int r = 0; r < 8; ++r) c[r] = 0.f;
      c = gemm_acc<8>(c, s_gl, 256, Wcpk, wave);
      c = gemm_acc<16>(c, s_ctx, 512, Wspk, wave);
      store_c(s_q, 256, wave, c);
    }
    __syncthreads();

    // ---- h1 = relu(q @ W1 + b1)
    {
      const float bias = b1g[wave * 16 + m16];
      v8f c;
#pragma unroll
      for (int r = 0; r < 8; ++r) c[r] = bias;
      c = gemm_acc<8>(c, s_q, 256, W1pk, wave);
#pragma unroll
      for (int r = 0; r < 8; ++r) c[r] = fmaxf(c[r], 0.f);
      store_c(s_h1, 256, wave, c);
    }
    __syncthreads();

    // ---- q2 = q + h1 @ W2 + b2  (residual)
    {
      const float bias = b2g[wave * 16 + m16];
      v8f c;
#pragma unroll
      for (int r = 0; r < 8; ++r)
        c[r] = s_q[(h16 * 8 + r) * 256 + wave * 16 + m16] + bias;
      c = gemm_acc<8>(c, s_h1, 256, W2pk, wave);
      store_c(s_q2, 256, wave, c);
    }
    __syncthreads();

    // ---- MHA glimpse, one batch element per wave, all via WMMA ------------
    {
      const int b = wave;
      // compat(8x128) = blockdiag(q2[b]) @ K^T : row h fed by k-tile kt==h
      const float* q2v = s_q2 + b * 256;
      const __bf16* Kb = K16 + (size_t)(bb0 + b) * (128 * 256);
      v8f acc[8];
#pragma unroll
      for (int nt = 0; nt < 8; ++nt)
#pragma unroll
        for (int r = 0; r < 8; ++r) acc[nt][r] = 0.f;
#pragma unroll
      for (int kt = 0; kt < 8; ++kt) {
        v16bf a = frag_a_onerow(q2v, kt, kt);
#pragma unroll
        for (int nt = 0; nt < 8; ++nt)
          acc[nt] = WMMA_BF16(a, frag_bT(Kb, 256, nt, kt), acc[nt]);
      }
      // scale + visited mask; values for head h live in acc[nt][h], lanes<16
      float x[8][8];
#pragma unroll
      for (int nt = 0; nt < 8; ++nt) {
        const int n = nt * 16 + m16;
        const bool msk = (s_vis[b * 4 + (n >> 5)] >> (n & 31)) & 1u;
#pragma unroll
        for (int r = 0; r < 8; ++r)
          x[nt][r] = msk ? -1e9f : acc[nt][r] * 0.17677669529663687f;
      }
      // per-head softmax over 128 entries (8 regs x lanes 0..15)
      __bf16* at = s_at + b * 1024;
#pragma unroll
      for (int r = 0; r < 8; ++r) {
        float mx = x[0][r];
#pragma unroll
        for (int nt = 1; nt < 8; ++nt) mx = fmaxf(mx, x[nt][r]);
#pragma unroll
        for (int off = 8; off; off >>= 1) mx = fmaxf(mx, __shfl_xor(mx, off));
        float sum = 0.f;
#pragma unroll
        for (int nt = 0; nt < 8; ++nt) sum += expf(x[nt][r] - mx);
#pragma unroll
        for (int off = 8; off; off >>= 1) sum += __shfl_xor(sum, off);
        const float inv = 1.f / sum;
        if (lane < 16) {
#pragma unroll
          for (int nt = 0; nt < 8; ++nt)
            at[r * 128 + nt * 16 + m16] = (__bf16)(expf(x[nt][r] - mx) * inv);
        }
      }
      // heads = attn(rows 0..7) @ V, via Vt (contiguous B fragments);
      // keep only block-diagonal output: col tile nt needs row nt>>1.
      const __bf16* Vb = Vt16 + (size_t)(bb0 + b) * (256 * 128);
#pragma unroll
      for (int nt = 0; nt < 16; ++nt) {
        v8f c;
#pragma unroll
        for (int r = 0; r < 8; ++r) c[r] = 0.f;
#pragma unroll
        for (int kt = 0; kt < 4; ++kt)
          c = WMMA_BF16(frag_a_rows8(at, kt), frag_bT(Vb, 128, nt, kt), c);
        if (lane < 16) s_hd[b * 256 + nt * 16 + m16] = c[nt >> 1];
      }
    }
    __syncthreads();

    // ---- new glimpse = heads @ Wo
    {
      v8f c;
#pragma unroll
      for (int r = 0; r < 8; ++r) c[r] = 0.f;
      c = gemm_acc<8>(c, s_hd, 256, Wopk, wave);
      store_c(s_gl, 256, wave, c);
    }
    __syncthreads();

    // ---- pointer logits via WMMA, greedy select, state update -------------
    {
      const int b = wave;
      const float* gl = s_gl + b * 256;
      const __bf16* Lb = L16 + (size_t)(bb0 + b) * (128 * 256);
      v8f acc[8];
#pragma unroll
      for (int nt = 0; nt < 8; ++nt)
#pragma unroll
        for (int r = 0; r < 8; ++r) acc[nt][r] = 0.f;
#pragma unroll
      for (int kt = 0; kt < 8; ++kt) {
        v16bf a = frag_a_onerow(gl, kt, 0);
#pragma unroll
        for (int nt = 0; nt < 8; ++nt)
          acc[nt] = WMMA_BF16(a, frag_bT(Lb, 256, nt, kt), acc[nt]);
      }
      if (lane < 16) {
#pragma unroll
        for (int nt = 0; nt < 8; ++nt) {
          const int n = nt * 16 + m16;
          float v = 10.f * tanhf(acc[nt][0] * (1.f / 16.f));
          if ((s_vis[b * 4 + (n >> 5)] >> (n & 31)) & 1u) v = -1e9f;
          s_lg[b * 128 + n] = v;
        }
      }
      // greedy argmax + log-softmax over 128 (same wave wrote s_lg; LDS ops
      // are in-order within a wave)
      float lg[4];
#pragma unroll
      for (int c4 = 0; c4 < 4; ++c4) lg[c4] = s_lg[b * 128 + lane + c4 * 32];
      float mx = lg[0];
      int am = lane;
#pragma unroll
      for (int c4 = 1; c4 < 4; ++c4) {
        const int n = lane + c4 * 32;
        if (lg[c4] > mx) { mx = lg[c4]; am = n; }
      }
#pragma unroll
      for (int off = 16; off; off >>= 1) {
        const float om = __shfl_xor(mx, off);
        const int oa = __shfl_xor(am, off);
        if (om > mx || (om == mx && oa < am)) { mx = om; am = oa; }
      }
      float s = 0.f;
#pragma unroll
      for (int c4 = 0; c4 < 4; ++c4) s += expf(lg[c4] - mx);
#pragma unroll
      for (int off = 16; off; off >>= 1) s += __shfl_xor(s, off);
      if (lane == 0) {
        const float lse = mx + logf(s);
        s_ll[b] += mx - lse;  // log_p at greedy argmax
        s_sel[b] = am;
        s_vis[b * 4 + (am >> 5)] |= 1u << (am & 31);
        const float* co = coords + ((size_t)(bb0 + b) * 128 + am) * 2;
        const float cx = co[0], cy = co[1];
        if (step == 0) {
          s_first[b] = am; s_fco[b * 2] = cx; s_fco[b * 2 + 1] = cy;
        } else {
          const float dx = cx - s_pco[b * 2], dy = cy - s_pco[b * 2 + 1];
          s_cost[b] += sqrtf(dx * dx + dy * dy);
        }
        s_prev[b] = am; s_pco[b * 2] = cx; s_pco[b * 2 + 1] = cy;
      }
    }
    __syncthreads();
  }

  if (tid < 16) {
    const int b = tid;
    const float dx = s_pco[b * 2] - s_fco[b * 2];
    const float dy = s_pco[b * 2 + 1] - s_fco[b * 2 + 1];
    out[bb0 + b] = s_cost[b] + sqrtf(dx * dx + dy * dy);  // tour cost
    out[1024 + bb0 + b] = s_ll[b];                        // log-likelihood
  }
}

// ---------------------------------------------------------------------------
// Host launcher
// ---------------------------------------------------------------------------
extern "C" void kernel_launch(void* const* d_in, const int* in_sizes, int n_in,
                              void* d_out, int out_size, void* d_ws,
                              size_t ws_size, hipStream_t stream) {
  (void)in_sizes; (void)n_in; (void)out_size; (void)ws_size;
  const float* coords = (const float*)d_in[0];
  const float* E = (const float*)d_in[1];
  const float* Wk = (const float*)d_in[2];
  const float* Wv = (const float*)d_in[3];
  const float* Wlk = (const float*)d_in[4];
  const float* Wo = (const float*)d_in[5];
  const float* Wc = (const float*)d_in[6];
  const float* Ws = (const float*)d_in[7];
  const float* Wph = (const float*)d_in[8];
  const float* W1 = (const float*)d_in[9];
  const float* b1 = (const float*)d_in[10];
  const float* W2 = (const float*)d_in[11];
  const float* b2 = (const float*)d_in[12];
  float* out = (float*)d_out;

  const size_t M = 131072;  // B*N rows
  uintptr_t p = (uintptr_t)d_ws;
  auto take = [&](size_t bytes) -> uintptr_t {
    p = (p + 255) & ~(uintptr_t)255;
    uintptr_t r = p; p += bytes; return r;
  };
  __bf16* Epk  = (__bf16*)take(M * 256 * 2);
  __bf16* Wkpk = (__bf16*)take(256 * 256 * 2);
  __bf16* Wvpk = (__bf16*)take(256 * 256 * 2);
  __bf16* Wlpk = (__bf16*)take(256 * 256 * 2);
  __bf16* Wcpk = (__bf16*)take(256 * 256 * 2);
  __bf16* Wspk = (__bf16*)take(512 * 256 * 2);
  __bf16* W1pk = (__bf16*)take(256 * 256 * 2);
  __bf16* W2pk = (__bf16*)take(256 * 256 * 2);
  __bf16* Wopk = (__bf16*)take(256 * 256 * 2);
  __bf16* K16  = (__bf16*)take(M * 256 * 2);  // row-major [B*N][256]
  __bf16* Vt16 = (__bf16*)take(M * 256 * 2);  // per-batch transposed [B][256][128]
  __bf16* L16  = (__bf16*)take(M * 256 * 2);  // row-major [B*N][256]

  // 1) pack embeddings into A-fragment bf16 tiles
  k_pack_a<<<(int)(M / 16 * 8 * 32 / 256), 256, 0, stream>>>(E, Epk, 256,
                                                             (int)(M / 16));
  // 2) pack all weights into B-fragment bf16 tiles
  auto packB = [&](const float* W, __bf16* o, int K, int N) {
    const int total = (K >> 5) * (N >> 4) * 32;
    k_pack_b<<<(total + 255) / 256, 256, 0, stream>>>(W, o, K, N);
  };
  packB(Wk, Wkpk, 256, 256);
  packB(Wv, Wvpk, 256, 256);
  packB(Wlk, Wlpk, 256, 256);
  packB(Wc, Wcpk, 256, 256);
  packB(Ws, Wspk, 512, 256);
  packB(W1, W1pk, 256, 256);
  packB(W2, W2pk, 256, 256);
  packB(Wo, Wopk, 256, 256);

  // 3) K / V^T / logitK projections via bf16 WMMA (f32 accumulate)
  k_gemm_kvl<<<dim3(8192, 3), 256, 0, stream>>>(Epk, Wkpk, Wvpk, Wlpk, K16,
                                                Vt16, L16);

  // 4) persistent greedy decode: 64 blocks x 16 batch elements each
  k_decode<<<64, 512, DEC_SMEM_BYTES, stream>>>(coords, E, Wph, b1, b2, K16,
                                                Vt16, L16, Wcpk, Wspk, W1pk,
                                                W2pk, Wopk, out);
}
